// TransformerBlock_51591147159561
// MI455X (gfx1250) — compile-verified, matrix operand order checked by reading, not run on hardware
//
#include <hip/hip_runtime.h>
#include <hip/hip_bf16.h>
#include <math.h>

// ---------------------------------------------------------------------------
// Problem constants (from reference): B=2, N=1024, D=512, H=8, d=64, HID=2048
// Key algebraic simplification: the mean over the 4 sign-flip frames kills the
// PCA terms exactly, so attn bias = w_edge[2,h] * ||coords[i]-coords[j]||.
// All GEMMs run as bf16 WMMA (v_wmma_f32_16x16x32_bf16) with f32 accumulate.
// GEMM v3: 32x64 wave tile (2 A-frags reuse each B-frag -> 1.5 b128 loads per
// wmma) with ONLY constant-indexed fragment arrays so everything stays in
// VGPRs (round-2 runtime-indexed double-buffer caused scratch spills).
// ---------------------------------------------------------------------------
#define D_MODEL  512
#define N_HEADS  8
#define HEAD_DIM 64
#define HIDDEN   2048
#define BATCH    2
#define SEQ      1024
#define ROWS     (BATCH * SEQ)      // 2048
#define BH       (BATCH * N_HEADS)  // 16

typedef float  v8f   __attribute__((ext_vector_type(8)));
typedef __bf16 v16bf __attribute__((ext_vector_type(16)));

union FragA { v16bf v; unsigned u[8]; };
union FragC { v8f   v; float    f[8]; };

__device__ __forceinline__ unsigned short f2bf(float x) {
  unsigned u = __float_as_uint(x);
  u += 0x7fffu + ((u >> 16) & 1u);   // round-to-nearest-even
  return (unsigned short)(u >> 16);
}

// K-offset of the bf16 pair held in A/B fragment VGPR `vr` for this half-wave
// (16-bit A-matrix 16x32 layout, CDNA5 ISA 7.12.2).
__device__ __forceinline__ int frag_k(int vr, int half) {
  return (vr < 4 ? 2 * vr : 16 + 2 * (vr - 4)) + (half ? 8 : 0);
}

// ---------------------------------------------------------------------------
// Weight prep: fp32 (K,N) row-major -> bf16 (N,K) "B-transposed" so WMMA
// B-fragment loads are contiguous 32-bit loads of K-pairs.
// ---------------------------------------------------------------------------
__global__ void wt_bf16(const float* __restrict__ w, unsigned short* __restrict__ wt,
                        int K, int N) {
  int idx = blockIdx.x * blockDim.x + threadIdx.x;
  if (idx >= K * N) return;
  int n = idx % N, k = idx / N;
  wt[(size_t)n * K + k] = f2bf(w[idx]);
}

// ---------------------------------------------------------------------------
// Row LayerNorm (C=512), fp32 stats, bf16 output. One 256-thread block / row.
// ---------------------------------------------------------------------------
__global__ __launch_bounds__(256) void ln_rows(const float* __restrict__ x,
                                               const float* __restrict__ g,
                                               const float* __restrict__ bb,
                                               unsigned short* __restrict__ out) {
  int row = blockIdx.x, tid = threadIdx.x;
  const float* xr = x + (size_t)row * D_MODEL;
  float v0 = xr[tid], v1 = xr[tid + 256];
  __shared__ float s1[256], s2[256];
  s1[tid] = v0 + v1;
  s2[tid] = v0 * v0 + v1 * v1;
  __syncthreads();
  for (int off = 128; off > 0; off >>= 1) {
    if (tid < off) { s1[tid] += s1[tid + off]; s2[tid] += s2[tid + off]; }
    __syncthreads();
  }
  float mu  = s1[0] * (1.0f / D_MODEL);
  float var = s2[0] * (1.0f / D_MODEL) - mu * mu;
  float rs  = rsqrtf(var + 1e-5f);
  unsigned short* orow = out + (size_t)row * D_MODEL;
  orow[tid]       = f2bf((v0 - mu) * rs * g[tid] + bb[tid]);
  orow[tid + 256] = f2bf((v1 - mu) * rs * g[tid + 256] + bb[tid + 256]);
}

// ---------------------------------------------------------------------------
// bf16 WMMA GEMM: C = A(MxK) @ B(KxN) [+bias] [gelu] [+residual].
// A: bf16 row-major; Bt: bf16 (N,K). Wave -> 32x64 tile: per 32-wide K step,
// 12 global_load_b128 feed 8 v_wmma (B-frags shared by 2 row-frags).
// WG (4 waves) -> 128x64 tile. Fragments use constant indices only.
// ---------------------------------------------------------------------------
__global__ __launch_bounds__(128) void gemm_bf16(
    const unsigned short* __restrict__ A,
    const unsigned short* __restrict__ Bt,
    const float* __restrict__ bias,
    const float* __restrict__ residual,
    float* __restrict__ outF,
    unsigned short* __restrict__ outB,
    int M, int N, int K, int gelu) {
  int wave = threadIdx.x >> 5;             // 0..3
  int lane = threadIdx.x & 31;
  int half = lane >> 4, lr = lane & 15;
  int row0 = blockIdx.y * 128 + wave * 32;
  int col0 = blockIdx.x * 64;

  FragC acc[2][4];
#pragma unroll
  for (int r = 0; r < 2; ++r)
#pragma unroll
    for (int t = 0; t < 4; ++t)
#pragma unroll
      for (int i = 0; i < 8; ++i) acc[r][t].f[i] = 0.f;

  const unsigned short* arow0 = A + (size_t)(row0 + lr) * K;
  const unsigned short* arow1 = A + (size_t)(row0 + 16 + lr) * K;

  for (int kk = 0; kk < K; kk += 32) {
    FragA a0, a1, b[4];
#pragma unroll
    for (int vr = 0; vr < 8; ++vr) {
      int kit = kk + frag_k(vr, half);
      a0.u[vr] = *(const unsigned*)(arow0 + kit);
      a1.u[vr] = *(const unsigned*)(arow1 + kit);
    }
#pragma unroll
    for (int t = 0; t < 4; ++t) {
      const unsigned short* bcol = Bt + (size_t)(col0 + t * 16 + lr) * K + kk;
#pragma unroll
      for (int vr = 0; vr < 8; ++vr)
        b[t].u[vr] = *(const unsigned*)(bcol + frag_k(vr, half));
    }
#pragma unroll
    for (int t = 0; t < 4; ++t) {
      acc[0][t].v = __builtin_amdgcn_wmma_f32_16x16x32_bf16(
          false, a0.v, false, b[t].v, (short)0, acc[0][t].v, false, false);
      acc[1][t].v = __builtin_amdgcn_wmma_f32_16x16x32_bf16(
          false, a1.v, false, b[t].v, (short)0, acc[1][t].v, false, false);
    }
  }

#pragma unroll
  for (int r = 0; r < 2; ++r)
#pragma unroll
    for (int t = 0; t < 4; ++t) {
      int col = col0 + t * 16 + lr;
      float bv = bias ? bias[col] : 0.f;
#pragma unroll
      for (int vr = 0; vr < 8; ++vr) {
        int row = row0 + r * 16 + vr + half * 8;   // C-layout: VGPR r -> M=r/r+8
        float v = acc[r][t].f[vr] + bv;
        if (gelu) v = 0.5f * v * (1.f + erff(v * 0.70710678118654752f));
        if (residual) v += residual[(size_t)row * N + col];
        if (outF) outF[(size_t)row * N + col] = v;
        if (outB) outB[(size_t)row * N + col] = f2bf(v);
      }
    }
}

// ---------------------------------------------------------------------------
// Split fp32 qkv (ROWS, 3*D) into per-head bf16 buffers:
//   q: (BH,N,64) pre-scaled by d^-0.5; k: (BH,N,64); vT: (BH,64,N).
// Layouts chosen so every WMMA fragment load in attention is contiguous.
// ---------------------------------------------------------------------------
__global__ void pack_qkv(const float* __restrict__ qkv,
                         unsigned short* __restrict__ q,
                         unsigned short* __restrict__ k,
                         unsigned short* __restrict__ vT) {
  int idx = blockIdx.x * blockDim.x + threadIdx.x;
  if (idx >= ROWS * D_MODEL) return;
  int bn = idx / D_MODEL, c = idx % D_MODEL;
  int b = bn / SEQ, n = bn % SEQ;
  int h = c / HEAD_DIM, dd = c % HEAD_DIM;
  const float* p = qkv + (size_t)bn * (3 * D_MODEL);
  int bh = b * N_HEADS + h;
  q[((size_t)bh * SEQ + n) * HEAD_DIM + dd]  = f2bf(p[c] * 0.125f);  // 64^-0.5
  k[((size_t)bh * SEQ + n) * HEAD_DIM + dd]  = f2bf(p[D_MODEL + c]);
  vT[((size_t)bh * HEAD_DIM + dd) * SEQ + n] = f2bf(p[2 * D_MODEL + c]);
}

// ---------------------------------------------------------------------------
// Flash attention: 1 wave per 16-row Q tile. Per 32-key chunk:
//   2 S-tiles (4 wmma) -> +radial bias/mask -> online softmax (shfl_xor over
//   the half-wave that owns each row) -> P bounced through wave-private LDS
//   into A-fragment layout -> 4 wmma accumulating O over 4 dim-tiles.
// ---------------------------------------------------------------------------
__global__ __launch_bounds__(256) void attn_fa(
    const unsigned short* __restrict__ q,    // (BH,N,64), pre-scaled
    const unsigned short* __restrict__ k,    // (BH,N,64)
    const unsigned short* __restrict__ vT,   // (BH,64,N)
    const float* __restrict__ coords,        // (B,N,2)
    const unsigned char* __restrict__ mask,  // (B,N,N) bool
    const float* __restrict__ w_edge,        // (3,H)
    unsigned short* __restrict__ o) {        // (ROWS, D) bf16
  __shared__ unsigned short pshm[8][16 * 32];
  int wave = threadIdx.x >> 5;
  int lane = threadIdx.x & 31;
  int half = lane >> 4, lr = lane & 15;
  int gw = blockIdx.x * 8 + wave;
  int bh = gw / (SEQ / 16);
  int row0 = (gw % (SEQ / 16)) * 16;
  int b = bh / N_HEADS, h = bh % N_HEADS;
  float wb = w_edge[2 * N_HEADS + h];  // only surviving frame-avg term

  FragA aq0, aq1;
  {
    const unsigned short* qrow = q + ((size_t)bh * SEQ + row0 + lr) * HEAD_DIM;
#pragma unroll
    for (int vr = 0; vr < 8; ++vr) {
      int kit = frag_k(vr, half);
      aq0.u[vr] = *(const unsigned*)(qrow + kit);
      aq1.u[vr] = *(const unsigned*)(qrow + 32 + kit);
    }
  }
  float cix[8], ciy[8];
#pragma unroll
  for (int vr = 0; vr < 8; ++vr) {
    int i = row0 + vr + half * 8;
    cix[vr] = coords[((size_t)b * SEQ + i) * 2 + 0];
    ciy[vr] = coords[((size_t)b * SEQ + i) * 2 + 1];
  }

  float m[8], l[8];
  FragC oacc[4];
#pragma unroll
  for (int vr = 0; vr < 8; ++vr) { m[vr] = -3e38f; l[vr] = 0.f; }
#pragma unroll
  for (int t = 0; t < 4; ++t)
#pragma unroll
    for (int i = 0; i < 8; ++i) oacc[t].f[i] = 0.f;

  const unsigned char* mbase = mask + (size_t)b * SEQ * SEQ;

  for (int ch = 0; ch < SEQ / 32; ++ch) {
    int jbase = ch * 32;
    FragC s[2];
#pragma unroll
    for (int t2 = 0; t2 < 2; ++t2) {
      int jb = jbase + t2 * 16;
      FragA bk0, bk1;
      const unsigned short* krow = k + ((size_t)bh * SEQ + jb + lr) * HEAD_DIM;
#pragma unroll
      for (int vr = 0; vr < 8; ++vr) {
        int kit = frag_k(vr, half);
        bk0.u[vr] = *(const unsigned*)(krow + kit);
        bk1.u[vr] = *(const unsigned*)(krow + 32 + kit);
      }
#pragma unroll
      for (int i = 0; i < 8; ++i) s[t2].f[i] = 0.f;
      s[t2].v = __builtin_amdgcn_wmma_f32_16x16x32_bf16(
          false, aq0.v, false, bk0.v, (short)0, s[t2].v, false, false);
      s[t2].v = __builtin_amdgcn_wmma_f32_16x16x32_bf16(
          false, aq1.v, false, bk1.v, (short)0, s[t2].v, false, false);
      // radial-distance bias + padding mask (C layout: col=lr, row=vr+half*8)
      int j = jb + lr;
      float cjx = coords[((size_t)b * SEQ + j) * 2 + 0];
      float cjy = coords[((size_t)b * SEQ + j) * 2 + 1];
#pragma unroll
      for (int vr = 0; vr < 8; ++vr) {
        int i2 = row0 + vr + half * 8;
        float dx = cix[vr] - cjx, dy = ciy[vr] - cjy;
        float sv = s[t2].f[vr] + wb * sqrtf(dx * dx + dy * dy);
        if (mbase[(size_t)i2 * SEQ + j]) sv = -1e9f;
        s[t2].f[vr] = sv;
      }
    }
    // online softmax; each row lives across the 16 lanes of one half-wave
#pragma unroll
    for (int vr = 0; vr < 8; ++vr) {
      float mx = fmaxf(s[0].f[vr], s[1].f[vr]);
#pragma unroll
      for (int off = 1; off < 16; off <<= 1) mx = fmaxf(mx, __shfl_xor(mx, off, 32));
      float mnew = fmaxf(m[vr], mx);
      float corr = __expf(m[vr] - mnew);
      float p0 = __expf(s[0].f[vr] - mnew);
      float p1 = __expf(s[1].f[vr] - mnew);
      float rs = p0 + p1;
#pragma unroll
      for (int off = 1; off < 16; off <<= 1) rs += __shfl_xor(rs, off, 32);
      l[vr] = l[vr] * corr + rs;
      m[vr] = mnew;
#pragma unroll
      for (int t = 0; t < 4; ++t) oacc[t].f[vr] *= corr;
      int prow = vr + half * 8;                 // C-layout row -> LDS row
      pshm[wave][prow * 32 + lr]      = f2bf(p0);
      pshm[wave][prow * 32 + 16 + lr] = f2bf(p1);
    }
    asm volatile("s_wait_dscnt 0x0" ::: "memory");  // wave-private LDS bounce
    FragA ap;  // re-read P in A-fragment layout (row = lr, K = key offset)
#pragma unroll
    for (int vr = 0; vr < 8; ++vr)
      ap.u[vr] = *(const unsigned*)&pshm[wave][lr * 32 + frag_k(vr, half)];
#pragma unroll
    for (int t = 0; t < 4; ++t) {
      FragA bv;  // B = V chunk: lane col = dim, VGPR pairs = consecutive keys
      const unsigned short* vrow =
          vT + ((size_t)bh * HEAD_DIM + t * 16 + lr) * SEQ + jbase;
#pragma unroll
      for (int vr = 0; vr < 8; ++vr)
        bv.u[vr] = *(const unsigned*)(vrow + frag_k(vr, half));
      oacc[t].v = __builtin_amdgcn_wmma_f32_16x16x32_bf16(
          false, ap.v, false, bv.v, (short)0, oacc[t].v, false, false);
    }
  }

#pragma unroll
  for (int t = 0; t < 4; ++t)
#pragma unroll
    for (int vr = 0; vr < 8; ++vr) {
      int i2 = row0 + vr + half * 8;
      float val = oacc[t].f[vr] / l[vr];
      o[(size_t)(b * SEQ + i2) * D_MODEL + h * HEAD_DIM + t * 16 + lr] = f2bf(val);
    }
}

// ---------------------------------------------------------------------------
extern "C" void kernel_launch(void* const* d_in, const int* in_sizes, int n_in,
                              void* d_out, int out_size, void* d_ws, size_t ws_size,
                              hipStream_t stream) {
  (void)in_sizes; (void)n_in; (void)out_size; (void)ws_size;
  const float* x      = (const float*)d_in[0];
  const float* coords = (const float*)d_in[1];
  const unsigned char* pmask = (const unsigned char*)d_in[2];
  const float* ln1_g = (const float*)d_in[3];
  const float* ln1_b = (const float*)d_in[4];
  const float* w_qkv = (const float*)d_in[5];
  const float* b_qkv = (const float*)d_in[6];
  const float* w_edge = (const float*)d_in[7];
  const float* w_out = (const float*)d_in[8];
  const float* b_out = (const float*)d_in[9];
  const float* ln2_g = (const float*)d_in[10];
  const float* ln2_b = (const float*)d_in[11];
  const float* w1 = (const float*)d_in[12];
  const float* b1 = (const float*)d_in[13];
  const float* w2 = (const float*)d_in[14];
  const float* b2 = (const float*)d_in[15];
  float* out = (float*)d_out;

  char* ws = (char*)d_ws;
  size_t off = 0;
  auto alloc = [&](size_t bytes) -> char* {
    char* p = ws + off;
    off += (bytes + 255) & ~(size_t)255;
    return p;
  };
  unsigned short* wqkvT = (unsigned short*)alloc((size_t)3 * D_MODEL * D_MODEL * 2);
  unsigned short* woutT = (unsigned short*)alloc((size_t)D_MODEL * D_MODEL * 2);
  unsigned short* w1T   = (unsigned short*)alloc((size_t)D_MODEL * HIDDEN * 2);
  unsigned short* w2T   = (unsigned short*)alloc((size_t)HIDDEN * D_MODEL * 2);
  unsigned short* xn    = (unsigned short*)alloc((size_t)ROWS * D_MODEL * 2);
  float*          qkvF  = (float*)alloc((size_t)ROWS * 3 * D_MODEL * 4);
  unsigned short* qbf   = (unsigned short*)alloc((size_t)BH * SEQ * HEAD_DIM * 2);
  unsigned short* kbf   = (unsigned short*)alloc((size_t)BH * SEQ * HEAD_DIM * 2);
  unsigned short* vTbf  = (unsigned short*)alloc((size_t)BH * SEQ * HEAD_DIM * 2);
  unsigned short* obf   = (unsigned short*)alloc((size_t)ROWS * D_MODEL * 2);
  float*          x1    = (float*)alloc((size_t)ROWS * D_MODEL * 4);
  unsigned short* ybf   = (unsigned short*)alloc((size_t)ROWS * D_MODEL * 2);
  unsigned short* hbf   = (unsigned short*)alloc((size_t)ROWS * HIDDEN * 2);

  const int thr = 256;
  // 1) weights -> bf16 transposed
  wt_bf16<<<(D_MODEL * 3 * D_MODEL + thr - 1) / thr, thr, 0, stream>>>(w_qkv, wqkvT, D_MODEL, 3 * D_MODEL);
  wt_bf16<<<(D_MODEL * D_MODEL + thr - 1) / thr, thr, 0, stream>>>(w_out, woutT, D_MODEL, D_MODEL);
  wt_bf16<<<(D_MODEL * HIDDEN + thr - 1) / thr, thr, 0, stream>>>(w1, w1T, D_MODEL, HIDDEN);
  wt_bf16<<<(HIDDEN * D_MODEL + thr - 1) / thr, thr, 0, stream>>>(w2, w2T, HIDDEN, D_MODEL);
  // 2) LN1 -> bf16
  ln_rows<<<ROWS, 256, 0, stream>>>(x, ln1_g, ln1_b, xn);
  // 3) QKV GEMM (2048x1536x512)
  gemm_bf16<<<dim3(3 * D_MODEL / 64, ROWS / 128), 128, 0, stream>>>(
      xn, wqkvT, b_qkv, nullptr, qkvF, nullptr, ROWS, 3 * D_MODEL, D_MODEL, 0);
  // 4) split/scale/transpose into attention layouts
  pack_qkv<<<(ROWS * D_MODEL + thr - 1) / thr, thr, 0, stream>>>(qkvF, qbf, kbf, vTbf);
  // 5) flash attention with radial bias
  attn_fa<<<(BH * SEQ / 16) / 8, 256, 0, stream>>>(qbf, kbf, vTbf, coords, pmask, w_edge, obf);
  // 6) out-proj + residual(x) -> x1
  gemm_bf16<<<dim3(D_MODEL / 64, ROWS / 128), 128, 0, stream>>>(
      obf, woutT, b_out, x, x1, nullptr, ROWS, D_MODEL, D_MODEL, 0);
  // 7) LN2 -> bf16
  ln_rows<<<ROWS, 256, 0, stream>>>(x1, ln2_g, ln2_b, ybf);
  // 8) MLP up + exact GELU -> bf16
  gemm_bf16<<<dim3(HIDDEN / 64, ROWS / 128), 128, 0, stream>>>(
      ybf, w1T, b1, nullptr, nullptr, hbf, ROWS, HIDDEN, D_MODEL, 1);
  // 9) MLP down + residual(x1) -> out
  gemm_bf16<<<dim3(D_MODEL / 64, ROWS / 128), 128, 0, stream>>>(
      hbf, w2T, b2, x1, out, nullptr, ROWS, D_MODEL, HIDDEN, 0);
}
